// GraphSAGEClassifier_74723841016089
// MI455X (gfx1250) — matrix operand order repaired and checked by reading, not compiled, not run on hardware
//
#include <hip/hip_runtime.h>
#include <hip/hip_bf16.h>

// ---------------------------------------------------------------------------
// GraphSAGE (2x SAGEConv mean-aggr + linear head) for MI455X / gfx1250.
// bf16 WMMA (v_wmma_f32_16x16x32_bf16) for the four big GEMMs, f32 atomic
// scatter-sum for the mean aggregation (L2-resident: 192MB L2 >> 128MB WS).
// ---------------------------------------------------------------------------

typedef __attribute__((ext_vector_type(16))) __bf16 v16bf;
typedef __attribute__((ext_vector_type(8)))  __bf16 v8bf;
typedef __attribute__((ext_vector_type(4)))  __bf16 v4bf;
typedef __attribute__((ext_vector_type(8)))  float  v8f;
typedef __attribute__((ext_vector_type(4)))  float  v4f;

#define N_NODES 50000
#define N_EDGES 600000
#define IN_DIM  128
#define HID     256
#define OUTC    3

// ---- weight convert + transpose: Wt[n*K+k] = bf16(W[k*N+n]) ---------------
__global__ __launch_bounds__(256) void wt_transpose_kernel(
    const float* __restrict__ W, __bf16* __restrict__ Wt, int K, int N) {
  int t = blockIdx.x * blockDim.x + threadIdx.x;
  if (t >= K * N) return;
  int n = t / K, k = t - n * K;
  Wt[t] = (__bf16)W[k * N + n];
}

// ---- degree: deg[dst[e]] += 1 ---------------------------------------------
__global__ __launch_bounds__(256) void degree_kernel(
    const int* __restrict__ dst, float* __restrict__ deg, int E) {
  int e = blockIdx.x * blockDim.x + threadIdx.x;
  if (e >= E) return;
  atomicAdd(&deg[dst[e]], 1.0f);
}

// ---- scatter-sum (f32 source), 4 channels / thread ------------------------
__global__ __launch_bounds__(256) void scatter_f32_kernel(
    const float* __restrict__ x, const int* __restrict__ src,
    const int* __restrict__ dst, float* __restrict__ agg, int E, int D) {
  int t = blockIdx.x * blockDim.x + threadIdx.x;
  int per = D >> 2;
  if (t >= E * per) return;
  int e = t / per, c = (t - e * per) << 2;
  int s = src[e], d = dst[e];
  v4f v = *(const v4f*)(x + (size_t)s * D + c);
  float* o = agg + (size_t)d * D + c;
  atomicAdd(o + 0, v.x); atomicAdd(o + 1, v.y);
  atomicAdd(o + 2, v.z); atomicAdd(o + 3, v.w);
}

// ---- scatter-sum (bf16 source) --------------------------------------------
__global__ __launch_bounds__(256) void scatter_bf16_kernel(
    const __bf16* __restrict__ h, const int* __restrict__ src,
    const int* __restrict__ dst, float* __restrict__ agg, int E, int D) {
  int t = blockIdx.x * blockDim.x + threadIdx.x;
  int per = D >> 2;
  if (t >= E * per) return;
  int e = t / per, c = (t - e * per) << 2;
  int s = src[e], d = dst[e];
  v4bf v = *(const v4bf*)(h + (size_t)s * D + c);
  float* o = agg + (size_t)d * D + c;
  atomicAdd(o + 0, (float)v[0]); atomicAdd(o + 1, (float)v[1]);
  atomicAdd(o + 2, (float)v[2]); atomicAdd(o + 3, (float)v[3]);
}

// ---- mean = agg / max(deg,1), emit bf16 -----------------------------------
__global__ __launch_bounds__(256) void mean_kernel(
    const float* __restrict__ agg, const float* __restrict__ deg,
    __bf16* __restrict__ mean_bf, int N, int D) {
  int t = blockIdx.x * blockDim.x + threadIdx.x;
  if (t >= N * D) return;
  int n = t / D;
  float dg = deg[n];
  dg = dg > 1.0f ? dg : 1.0f;
  mean_bf[t] = (__bf16)(agg[t] / dg);
}

// ---------------------------------------------------------------------------
// Fused SAGE layer GEMM: out = relu(mean @ Wl + root @ Wr + bias), bf16 WMMA.
// One wave per 16x16 output tile; WlT/WrT are pre-transposed [Nout][K].
//
// A-operand (16x32 bf16, ISA 7.12.2): lane L holds row L&15; elems 0..15 are
// K = kh*16+i (kh = L>>4; this code's order, not the ISA table's) -> one
// contiguous 32B row segment. B-operand: lane L holds col L&15, K = kh*16+i
// -> one contiguous 32B segment of the transposed weight row.
// C/D (f32): lane L -> col L&15, VGPR r -> row r + 8*kh.
// ---------------------------------------------------------------------------
template <int K, bool ROOT_F32>
__global__ __launch_bounds__(128) void sage_gemm_kernel(
    const __bf16* __restrict__ meanA,  // [M,K] bf16 row-major
    const void*  __restrict__ rootA,   // [M,K] f32 (layer1) or bf16 (layer2)
    const __bf16* __restrict__ WlT,    // [Nout,K] bf16
    const __bf16* __restrict__ WrT,    // [Nout,K] bf16
    const float* __restrict__ bias,    // [Nout] f32
    __bf16* __restrict__ outp,         // [M,Nout] bf16
    int Nout) {
  const int wave = threadIdx.x >> 5;
  const int lane = threadIdx.x & 31;
  const int row  = lane & 15;           // A row / D col
  const int kh   = lane >> 4;
  const int m0   = blockIdx.x * 16;
  const int n0   = (blockIdx.y * 4 + wave) * 16;
  const int col  = lane & 15;

  const __bf16* mrow = meanA + (size_t)(m0 + row) * K;
  const __bf16* wl   = WlT + (size_t)(n0 + col) * K + kh * 16;
  const __bf16* wr   = WrT + (size_t)(n0 + col) * K + kh * 16;

  v8f acc = {};
#pragma unroll
  for (int k0 = 0; k0 < K; k0 += 32) {
    // A: aggregated-mean operand
    v16bf am;
    {
      v8bf lo = *(const v8bf*)(mrow + k0 + kh * 16);
      v8bf hi = *(const v8bf*)(mrow + k0 + kh * 16 + 8);
#pragma unroll
      for (int i = 0; i < 8; ++i) { am[i] = lo[i]; am[8 + i] = hi[i]; }
    }
    // A: root operand (x in f32 for layer1, h1 in bf16 for layer2)
    v16bf ar;
    if (ROOT_F32) {
      const float* rrow = (const float*)rootA + (size_t)(m0 + row) * K;
      v4f f0 = *(const v4f*)(rrow + k0 + kh * 16);
      v4f f1 = *(const v4f*)(rrow + k0 + kh * 16 + 4);
      v4f f2 = *(const v4f*)(rrow + k0 + kh * 16 + 8);
      v4f f3 = *(const v4f*)(rrow + k0 + kh * 16 + 12);
#pragma unroll
      for (int i = 0; i < 4; ++i) {
        ar[i]      = (__bf16)f0[i];
        ar[4 + i]  = (__bf16)f1[i];
        ar[8 + i]  = (__bf16)f2[i];
        ar[12 + i] = (__bf16)f3[i];
      }
    } else {
      const __bf16* rrow = (const __bf16*)rootA + (size_t)(m0 + row) * K;
      v8bf lo = *(const v8bf*)(rrow + k0 + kh * 16);
      v8bf hi = *(const v8bf*)(rrow + k0 + kh * 16 + 8);
#pragma unroll
      for (int i = 0; i < 8; ++i) { ar[i] = lo[i]; ar[8 + i] = hi[i]; }
    }
    // B: transposed weight rows (contiguous 32B per lane)
    v16bf bl = *(const v16bf*)(wl + k0);
    v16bf br = *(const v16bf*)(wr + k0);

    acc = __builtin_amdgcn_wmma_f32_16x16x32_bf16(false, am, false, bl,
                                                  (short)0, acc, false, false);
    acc = __builtin_amdgcn_wmma_f32_16x16x32_bf16(false, ar, false, br,
                                                  (short)0, acc, false, false);
  }
  const float b = bias[n0 + col];
#pragma unroll
  for (int r = 0; r < 8; ++r) {
    float v = acc[r] + b;
    v = v > 0.0f ? v : 0.0f;   // relu (both SAGE layers)
    outp[(size_t)(m0 + r + kh * 8) * Nout + (n0 + col)] = (__bf16)v;
  }
}

// ---- head: out[n,j] = sum_k h2[n,k]*Wh[k,j] + bh[j]  (OUTC=3, tiny) -------
__global__ __launch_bounds__(256) void head_kernel(
    const __bf16* __restrict__ h2, const float* __restrict__ Wh,
    const float* __restrict__ bh, float* __restrict__ out, int N) {
  int n = blockIdx.x * blockDim.x + threadIdx.x;
  if (n >= N) return;
  float a0 = bh[0], a1 = bh[1], a2 = bh[2];
  const __bf16* hr = h2 + (size_t)n * HID;
#pragma unroll 4
  for (int k = 0; k < HID; ++k) {
    float hv = (float)hr[k];
    a0 += hv * Wh[k * OUTC + 0];
    a1 += hv * Wh[k * OUTC + 1];
    a2 += hv * Wh[k * OUTC + 2];
  }
  out[(size_t)n * OUTC + 0] = a0;
  out[(size_t)n * OUTC + 1] = a1;
  out[(size_t)n * OUTC + 2] = a2;
}

static inline size_t ws_take(size_t& off, size_t bytes) {
  size_t o = off;
  off += (bytes + 255) & ~(size_t)255;
  return o;
}

extern "C" void kernel_launch(void* const* d_in, const int* in_sizes, int n_in,
                              void* d_out, int out_size, void* d_ws,
                              size_t ws_size, hipStream_t stream) {
  (void)in_sizes; (void)n_in; (void)out_size; (void)ws_size;
  const float* x   = (const float*)d_in[0];
  const int*   src = (const int*)d_in[1];            // edge_index row 0
  const int*   dst = (const int*)d_in[1] + N_EDGES;  // edge_index row 1
  const float* W1l = (const float*)d_in[2];
  const float* b1  = (const float*)d_in[3];
  const float* W1r = (const float*)d_in[4];
  const float* W2l = (const float*)d_in[5];
  const float* b2  = (const float*)d_in[6];
  const float* W2r = (const float*)d_in[7];
  const float* Wh  = (const float*)d_in[8];
  const float* bh  = (const float*)d_in[9];
  float* out = (float*)d_out;

  char* ws = (char*)d_ws;
  size_t off = 0;
  float*  deg     = (float*)(ws + ws_take(off, (size_t)N_NODES * 4));
  float*  agg     = (float*)(ws + ws_take(off, (size_t)N_NODES * HID * 4));
  __bf16* mean_bf = (__bf16*)(ws + ws_take(off, (size_t)N_NODES * HID * 2));
  __bf16* h1      = (__bf16*)(ws + ws_take(off, (size_t)N_NODES * HID * 2));
  __bf16* h2      = (__bf16*)(ws + ws_take(off, (size_t)N_NODES * HID * 2));
  __bf16* W1lT    = (__bf16*)(ws + ws_take(off, (size_t)IN_DIM * HID * 2));
  __bf16* W1rT    = (__bf16*)(ws + ws_take(off, (size_t)IN_DIM * HID * 2));
  __bf16* W2lT    = (__bf16*)(ws + ws_take(off, (size_t)HID * HID * 2));
  __bf16* W2rT    = (__bf16*)(ws + ws_take(off, (size_t)HID * HID * 2));

  const int B = 256;

  // weights: convert + transpose to [N][K] bf16 (one-time per launch, tiny)
  hipLaunchKernelGGL(wt_transpose_kernel, dim3((IN_DIM * HID + B - 1) / B),
                     dim3(B), 0, stream, W1l, W1lT, IN_DIM, HID);
  hipLaunchKernelGGL(wt_transpose_kernel, dim3((IN_DIM * HID + B - 1) / B),
                     dim3(B), 0, stream, W1r, W1rT, IN_DIM, HID);
  hipLaunchKernelGGL(wt_transpose_kernel, dim3((HID * HID + B - 1) / B),
                     dim3(B), 0, stream, W2l, W2lT, HID, HID);
  hipLaunchKernelGGL(wt_transpose_kernel, dim3((HID * HID + B - 1) / B),
                     dim3(B), 0, stream, W2r, W2rT, HID, HID);

  // degree (shared by both layers)
  hipMemsetAsync(deg, 0, (size_t)N_NODES * 4, stream);
  hipLaunchKernelGGL(degree_kernel, dim3((N_EDGES + B - 1) / B), dim3(B), 0,
                     stream, dst, deg, N_EDGES);

  // ---- layer 1: mean-aggregate x, then relu(mean@W1l + x@W1r + b1) -> h1
  hipMemsetAsync(agg, 0, (size_t)N_NODES * IN_DIM * 4, stream);
  {
    int tot = N_EDGES * (IN_DIM / 4);
    hipLaunchKernelGGL(scatter_f32_kernel, dim3((tot + B - 1) / B), dim3(B), 0,
                       stream, x, src, dst, agg, N_EDGES, IN_DIM);
  }
  hipLaunchKernelGGL(mean_kernel, dim3((N_NODES * IN_DIM + B - 1) / B),
                     dim3(B), 0, stream, agg, deg, mean_bf, N_NODES, IN_DIM);
  hipLaunchKernelGGL((sage_gemm_kernel<IN_DIM, true>),
                     dim3(N_NODES / 16, HID / 64), dim3(128), 0, stream,
                     mean_bf, (const void*)x, W1lT, W1rT, b1, h1, HID);

  // ---- layer 2: mean-aggregate h1, then relu(mean@W2l + h1@W2r + b2) -> h2
  hipMemsetAsync(agg, 0, (size_t)N_NODES * HID * 4, stream);
  {
    int tot = N_EDGES * (HID / 4);
    hipLaunchKernelGGL(scatter_bf16_kernel, dim3((tot + B - 1) / B), dim3(B),
                       0, stream, h1, src, dst, agg, N_EDGES, HID);
  }
  hipLaunchKernelGGL(mean_kernel, dim3((N_NODES * HID + B - 1) / B), dim3(B),
                     0, stream, agg, deg, mean_bf, N_NODES, HID);
  hipLaunchKernelGGL((sage_gemm_kernel<HID, false>),
                     dim3(N_NODES / 16, HID / 64), dim3(128), 0, stream,
                     mean_bf, (const void*)h1, W2lT, W2rT, b2, h2, HID);

  // ---- head
  hipLaunchKernelGGL(head_kernel, dim3((N_NODES + B - 1) / B), dim3(B), 0,
                     stream, h2, Wh, bh, out, N_NODES);
}
